// Block_24799141167439
// MI455X (gfx1250) — matrix operand order repaired and checked by reading, not compiled, not run on hardware
//
#include <hip/hip_runtime.h>
#include <hip/hip_bf16.h>

typedef __bf16 bf16;
typedef __attribute__((ext_vector_type(16))) __bf16 v16bf;
typedef __attribute__((ext_vector_type(8)))  float  v8f;

#define NEG_INF (-__builtin_inff())

constexpr int Bc = 4;
constexpr int Sc = 2048;
constexpr int Dc = 768;
constexpr int Hc = 12;
constexpr int Mc = Bc * Sc;          // 8192 token rows
constexpr float QK_SCALE = 0.03608439182435161f;  // 1/sqrt(768)

// ---------------- helpers ----------------

union B16x16 { v16bf v; uint4 u[2]; };
union B16x8  { uint4 u; __bf16 h[8]; };

__device__ inline v8f vzero8() {
  v8f v;
#pragma unroll
  for (int i = 0; i < 8; ++i) v[i] = 0.0f;
  return v;
}

// load a v16bf WMMA fragment from two 16-byte-aligned 8xbf16 chunks
__device__ inline v16bf load_frag(const bf16* p0, const bf16* p1) {
  B16x16 t;
  t.u[0] = *(const uint4*)p0;
  t.u[1] = *(const uint4*)p1;
  return t.v;
}

// async global -> LDS copy of 16 bytes per lane (CDNA5 async path, ASYNCcnt)
__device__ inline void async_cp16(const bf16* gptr, const bf16* ldsptr) {
  unsigned lds = (unsigned)(uintptr_t)ldsptr;  // LDS byte address (low 32 bits)
  asm volatile("global_load_async_to_lds_b128 %0, %1, off"
               :: "v"(lds), "v"(gptr)
               : "memory");
}

// ---------------- fp32 [K][N] -> bf16 [N][K] transpose-cast ----------------
// 32x32 tiles, 256 threads (32x8)

__global__ __launch_bounds__(256)
void castT_kernel(const float* __restrict__ in, bf16* __restrict__ out,
                  int K, int N) {
  __shared__ float tile[32][33];
  const int n0 = blockIdx.x * 32;
  const int k0 = blockIdx.y * 32;
  const int tx = threadIdx.x & 31;
  const int ty = threadIdx.x >> 5;
#pragma unroll
  for (int i = 0; i < 32; i += 8)
    tile[ty + i][tx] = in[(size_t)(k0 + ty + i) * N + n0 + tx];
  __syncthreads();
#pragma unroll
  for (int i = 0; i < 32; i += 8)
    out[(size_t)(n0 + ty + i) * K + k0 + tx] = (bf16)tile[tx][ty + i];
}

// ---------------- LayerNorm: fp32 in, bf16 out (row = 768) ----------------

__global__ __launch_bounds__(256)
void ln_bf16_kernel(const float* __restrict__ x, const float* __restrict__ g,
                    const float* __restrict__ beta, bf16* __restrict__ out) {
  const int row = blockIdx.x;
  const int tid = threadIdx.x;
  const float* xr = x + (size_t)row * Dc;
  float v0 = xr[tid], v1 = xr[tid + 256], v2 = xr[tid + 512];

  __shared__ float red[512];
  red[tid]       = v0 + v1 + v2;
  red[tid + 256] = v0 * v0 + v1 * v1 + v2 * v2;
  __syncthreads();
  for (int off = 128; off > 0; off >>= 1) {
    if (tid < off) {
      red[tid]       += red[tid + off];
      red[256 + tid] += red[256 + tid + off];
    }
    __syncthreads();
  }
  const float mu   = red[0] * (1.0f / Dc);
  const float var  = red[256] * (1.0f / Dc) - mu * mu;
  const float rstd = rsqrtf(var + 1e-5f);

  bf16* orow = out + (size_t)row * Dc;
  orow[tid]       = (bf16)((v0 - mu) * rstd * g[tid]       + beta[tid]);
  orow[tid + 256] = (bf16)((v1 - mu) * rstd * g[tid + 256] + beta[tid + 256]);
  orow[tid + 512] = (bf16)((v2 - mu) * rstd * g[tid + 512] + beta[tid + 512]);
}

// ---------------- bf16 WMMA GEMM, async double-buffered ----------------
// C[M,N] = A[M,K](bf16,row-major) * WT[N,K](bf16,row-major, i.e. B^T)
// (+bias)(+res)(gelu). Block tile 128x128, 8 waves (4M x 2N), wave 32x64, BK=32.
// Tiles staged via GLOBAL_LOAD_ASYNC_TO_LDS_B128 with 2-deep pipelining.

template<bool BIAS, bool GELU, bool RES, bool OUTBF>
__global__ __launch_bounds__(256)
void gemm_bf16_kernel(const bf16* __restrict__ A, const bf16* __restrict__ WT,
                      const float* __restrict__ bias, const float* __restrict__ res,
                      void* __restrict__ Cout, int M, int N, int K) {
  __shared__ bf16 As[2][128 * 40];   // [row][k] padded stride 40
  __shared__ bf16 Bs[2][128 * 40];   // [n][k]   padded stride 40

  const int tid  = threadIdx.x;
  const int wid  = tid >> 5;
  const int lane = tid & 31;
  const int half = lane >> 4;
  const int l16  = lane & 15;
  const int wm   = wid >> 1;      // 0..3
  const int wn   = wid & 1;       // 0..1
  const int m0   = blockIdx.y * 128;
  const int n0   = blockIdx.x * 128;

  // per-thread async-copy coordinates: 512 16B chunks per tile
  const int ar0 = tid >> 2,           ac0 = (tid & 3) << 3;          // A chunk 0
  const int ar1 = (tid + 256) >> 2,   ac1 = ac0;                     // A chunk 1
  const int bn0 = ar0, bk0 = ac0, bn1 = ar1, bk1 = ac1;              // B mirrors A

  auto issue_tile = [&](int buf, int k0) {
    async_cp16(A + (size_t)(m0 + ar0) * K + k0 + ac0, &As[buf][ar0 * 40 + ac0]);
    async_cp16(A + (size_t)(m0 + ar1) * K + k0 + ac1, &As[buf][ar1 * 40 + ac1]);
    async_cp16(WT + (size_t)(n0 + bn0) * K + k0 + bk0, &Bs[buf][bn0 * 40 + bk0]);
    async_cp16(WT + (size_t)(n0 + bn1) * K + k0 + bk1, &Bs[buf][bn1 * 40 + bk1]);
  };

  v8f acc[2][4];
#pragma unroll
  for (int mi = 0; mi < 2; ++mi)
#pragma unroll
    for (int ni = 0; ni < 4; ++ni) acc[mi][ni] = vzero8();

  issue_tile(0, 0);                       // prologue
  int cur = 0;
  for (int k0 = 0; k0 < K; k0 += 32) {
    const bool has_next = (k0 + 32) < K;
    if (has_next) {
      issue_tile(cur ^ 1, k0 + 32);       // prefetch next tile into other buffer
      asm volatile("s_wait_asynccnt 0x4" ::: "memory");  // current tile landed
    } else {
      asm volatile("s_wait_asynccnt 0x0" ::: "memory");
    }
    __syncthreads();

    const bf16* as = As[cur];
    const bf16* bs = Bs[cur];
    v16bf af[2], bfr[4];
#pragma unroll
    for (int mi = 0; mi < 2; ++mi) {
      int ar = wm * 32 + mi * 16 + l16;
      af[mi] = load_frag(&as[ar * 40 + 8 * half], &as[ar * 40 + 16 + 8 * half]);
    }
#pragma unroll
    for (int ni = 0; ni < 4; ++ni) {
      int bc = wn * 64 + ni * 16 + l16;
      bfr[ni] = load_frag(&bs[bc * 40 + 8 * half], &bs[bc * 40 + 16 + 8 * half]);
    }
#pragma unroll
    for (int mi = 0; mi < 2; ++mi)
#pragma unroll
      for (int ni = 0; ni < 4; ++ni)
        acc[mi][ni] = __builtin_amdgcn_wmma_f32_16x16x32_bf16(
            false, af[mi], false, bfr[ni], (short)0, acc[mi][ni], false, false);
    __syncthreads();   // all waves done reading before this buffer is refilled
    cur ^= 1;
  }

  // epilogue: lane = column, VGPR r = row (r + 8*half)
#pragma unroll
  for (int mi = 0; mi < 2; ++mi)
#pragma unroll
    for (int ni = 0; ni < 4; ++ni)
#pragma unroll
      for (int r = 0; r < 8; ++r) {
        int grow = m0 + wm * 32 + mi * 16 + half * 8 + r;
        int gcol = n0 + wn * 64 + ni * 16 + l16;
        float v = acc[mi][ni][r];
        if (BIAS) v += bias[gcol];
        if (RES)  v += res[(size_t)grow * N + gcol];
        if (GELU) v = 0.5f * v * (1.0f + erff(v * 0.70710678118654752f));
        if (OUTBF) ((bf16*)Cout)[(size_t)grow * N + gcol] = (bf16)v;
        else       ((float*)Cout)[(size_t)grow * N + gcol] = v;
      }
}

// ---------------- fused causal flash attention ----------------
// qkv: bf16 [B,S,3D]; ctx out: bf16 [B,S,D]. hd=64.
// Block = 4 waves = 128 threads; 64 query rows per block (16 per wave);
// loop over 64-key blocks up to the diagonal.

__global__ __launch_bounds__(128)
void attn_fused_kernel(const bf16* __restrict__ qkv, bf16* __restrict__ ctx) {
  __shared__ bf16 Ks[64 * 72];        // [key][hd]
  __shared__ bf16 Vt[64 * 72];        // transposed: [hd][key]
  __shared__ bf16 Pw[4 * 16 * 72];    // per-wave P staging [16 q][64 key]

  const int qb = blockIdx.x;          // 0..31
  const int h  = blockIdx.y;
  const int b  = blockIdx.z;
  const int tid  = threadIdx.x;
  const int wid  = tid >> 5;
  const int lane = tid & 31;
  const int half = lane >> 4;
  const int l16  = lane & 15;
  const int q0   = qb * 64;

  // Q fragments (A-layout), kept in registers for the whole kernel
  const int qrow = q0 + wid * 16 + l16;
  const bf16* qp = qkv + (size_t)(b * Sc + qrow) * (3 * Dc) + h * 64;
  v16bf qf[2];
#pragma unroll
  for (int kk = 0; kk < 2; ++kk)
    qf[kk] = load_frag(qp + kk * 32 + 8 * half, qp + kk * 32 + 16 + 8 * half);

  float m_i[8], l_i[8];
  v8f O[4];
#pragma unroll
  for (int r = 0; r < 8; ++r) { m_i[r] = NEG_INF; l_i[r] = 0.0f; }
#pragma unroll
  for (int nt = 0; nt < 4; ++nt) O[nt] = vzero8();

  for (int kb = 0; kb <= qb; ++kb) {
    const bf16* kp = qkv + (size_t)(b * Sc + kb * 64) * (3 * Dc) + Dc + h * 64;
    const bf16* vp = kp + Dc;
    // cooperative K load + V transpose load: 512 chunks of 8 bf16
#pragma unroll
    for (int i = 0; i < 4; ++i) {
      int c = tid + i * 128;
      int key = c >> 3, hd8 = (c & 7) << 3;
      *(uint4*)(&Ks[key * 72 + hd8]) =
          *(const uint4*)(kp + (size_t)key * (3 * Dc) + hd8);
      B16x8 t;
      t.u = *(const uint4*)(vp + (size_t)key * (3 * Dc) + hd8);
#pragma unroll
      for (int j = 0; j < 8; ++j) Vt[(hd8 + j) * 72 + key] = t.h[j];
    }
    __syncthreads();

    // S = Q * K^T  (16 q x 64 keys per wave)
    v8f Sa[4];
#pragma unroll
    for (int nt = 0; nt < 4; ++nt) {
      Sa[nt] = vzero8();
#pragma unroll
      for (int kk = 0; kk < 2; ++kk) {
        const bf16* kr = &Ks[(nt * 16 + l16) * 72 + kk * 32 + 8 * half];
        v16bf bfr = load_frag(kr, kr + 16);
        Sa[nt] = __builtin_amdgcn_wmma_f32_16x16x32_bf16(
            false, qf[kk], false, bfr, (short)0, Sa[nt], false, false);
      }
    }

    float sv[4][8];
#pragma unroll
    for (int nt = 0; nt < 4; ++nt)
#pragma unroll
      for (int r = 0; r < 8; ++r) sv[nt][r] = Sa[nt][r] * QK_SCALE;

    if (kb == qb) {   // causal mask only needed on the diagonal block
#pragma unroll
      for (int nt = 0; nt < 4; ++nt) {
        int j = kb * 64 + nt * 16 + l16;
#pragma unroll
        for (int r = 0; r < 8; ++r) {
          int q = q0 + wid * 16 + half * 8 + r;
          sv[nt][r] = (j > q) ? NEG_INF : sv[nt][r];
        }
      }
    }

    // online softmax: row reductions across the 16-lane half (C-layout rows)
#pragma unroll
    for (int r = 0; r < 8; ++r) {
      float t = fmaxf(fmaxf(sv[0][r], sv[1][r]), fmaxf(sv[2][r], sv[3][r]));
#pragma unroll
      for (int off = 8; off >= 1; off >>= 1) t = fmaxf(t, __shfl_xor(t, off, 32));
      float mn = fmaxf(m_i[r], t);
      float corr = __expf(m_i[r] - mn);
      m_i[r] = mn;
      float ps = 0.0f;
#pragma unroll
      for (int nt = 0; nt < 4; ++nt) {
        sv[nt][r] = __expf(sv[nt][r] - mn);
        ps += sv[nt][r];
      }
#pragma unroll
      for (int off = 8; off >= 1; off >>= 1) ps += __shfl_xor(ps, off, 32);
      l_i[r] = l_i[r] * corr + ps;
#pragma unroll
      for (int nt = 0; nt < 4; ++nt) O[nt][r] *= corr;
    }

    // stage P (C-layout -> LDS row-major) then reload as A-fragments
    bf16* pw = &Pw[wid * 16 * 72];
#pragma unroll
    for (int nt = 0; nt < 4; ++nt)
#pragma unroll
      for (int r = 0; r < 8; ++r)
        pw[(half * 8 + r) * 72 + nt * 16 + l16] = (bf16)sv[nt][r];
    __asm__ volatile("s_wait_dscnt 0x0" ::: "memory");

#pragma unroll
    for (int kk = 0; kk < 2; ++kk) {
      const bf16* pr = pw + l16 * 72 + kk * 32;
      v16bf pf = load_frag(pr + 8 * half, pr + 16 + 8 * half);
#pragma unroll
      for (int nt = 0; nt < 4; ++nt) {
        const bf16* vr = &Vt[(nt * 16 + l16) * 72 + kk * 32 + 8 * half];
        v16bf vf = load_frag(vr, vr + 16);
        O[nt] = __builtin_amdgcn_wmma_f32_16x16x32_bf16(
            false, pf, false, vf, (short)0, O[nt], false, false);
      }
    }
    __syncthreads();   // K/V/P LDS reused next iteration
  }

  // normalize and write ctx (bf16) for the O-projection GEMM
#pragma unroll
  for (int nt = 0; nt < 4; ++nt)
#pragma unroll
    for (int r = 0; r < 8; ++r) {
      int q  = q0 + wid * 16 + half * 8 + r;
      int hd = nt * 16 + l16;
      ctx[(size_t)(b * Sc + q) * Dc + h * 64 + hd] = (bf16)(O[nt][r] / l_i[r]);
    }
}

// ---------------- launcher ----------------

extern "C" void kernel_launch(void* const* d_in, const int* in_sizes, int n_in,
                              void* d_out, int out_size, void* d_ws, size_t ws_size,
                              hipStream_t stream) {
  (void)in_sizes; (void)n_in; (void)out_size; (void)ws_size;

  const float* x    = (const float*)d_in[0];
  const float* ln1g = (const float*)d_in[1];
  const float* ln1b = (const float*)d_in[2];
  const float* Wqkv = (const float*)d_in[3];
  const float* bqkv = (const float*)d_in[4];
  const float* Wo   = (const float*)d_in[5];
  const float* bo   = (const float*)d_in[6];
  const float* ln2g = (const float*)d_in[7];
  const float* ln2b = (const float*)d_in[8];
  const float* W1   = (const float*)d_in[9];
  const float* b1   = (const float*)d_in[10];
  const float* W2   = (const float*)d_in[11];
  const float* b2   = (const float*)d_in[12];
  float* out = (float*)d_out;

  // workspace carve-up (all sizes multiples of 256 bytes)
  char* ws = (char*)d_ws;
  size_t off = 0;
  auto carve = [&](size_t bytes) { char* p = ws + off; off += bytes; return p; };
  bf16* wqkvT = (bf16*)carve((size_t)Dc * 3 * Dc * 2);   // [2304][768]
  bf16* woT   = (bf16*)carve((size_t)Dc * Dc * 2);       // [768][768]
  bf16* w1T   = (bf16*)carve((size_t)Dc * 4 * Dc * 2);   // [3072][768]
  bf16* w2T   = (bf16*)carve((size_t)4 * Dc * Dc * 2);   // [768][3072]
  bf16* xn1   = (bf16*)carve((size_t)Mc * Dc * 2);
  bf16* qkvb  = (bf16*)carve((size_t)Mc * 3 * Dc * 2);
  bf16* ctxb  = (bf16*)carve((size_t)Mc * Dc * 2);
  float* x1   = (float*)carve((size_t)Mc * Dc * 4);
  bf16* xn2   = (bf16*)carve((size_t)Mc * Dc * 2);
  bf16* h1    = (bf16*)carve((size_t)Mc * 4 * Dc * 2);

  // 1) weights -> bf16, transposed to [N][K] so GEMM tiles are async-copyable
  castT_kernel<<<dim3(3 * Dc / 32, Dc / 32), 256, 0, stream>>>(Wqkv, wqkvT, Dc, 3 * Dc);
  castT_kernel<<<dim3(Dc / 32, Dc / 32), 256, 0, stream>>>(Wo, woT, Dc, Dc);
  castT_kernel<<<dim3(4 * Dc / 32, Dc / 32), 256, 0, stream>>>(W1, w1T, Dc, 4 * Dc);
  castT_kernel<<<dim3(Dc / 32, 4 * Dc / 32), 256, 0, stream>>>(W2, w2T, 4 * Dc, Dc);

  // 2) LN1 -> bf16
  ln_bf16_kernel<<<Mc, 256, 0, stream>>>(x, ln1g, ln1b, xn1);

  // 3) QKV GEMM: [8192,768] x [768,2304] + bias -> bf16
  gemm_bf16_kernel<true, false, false, true>
      <<<dim3((3 * Dc) / 128, Mc / 128), 256, 0, stream>>>(
          xn1, wqkvT, bqkv, nullptr, qkvb, Mc, 3 * Dc, Dc);

  // 4) fused causal attention -> ctx bf16
  attn_fused_kernel<<<dim3(Sc / 64, Hc, Bc), 128, 0, stream>>>(qkvb, ctxb);

  // 5) O-proj + bias + residual(x) -> x1 fp32
  gemm_bf16_kernel<true, false, true, false>
      <<<dim3(Dc / 128, Mc / 128), 256, 0, stream>>>(
          ctxb, woT, bo, x, x1, Mc, Dc, Dc);

  // 6) LN2 -> bf16
  ln_bf16_kernel<<<Mc, 256, 0, stream>>>(x1, ln2g, ln2b, xn2);

  // 7) MLP1 + bias + exact GELU -> bf16
  gemm_bf16_kernel<true, true, false, true>
      <<<dim3((4 * Dc) / 128, Mc / 128), 256, 0, stream>>>(
          xn2, w1T, b1, nullptr, h1, Mc, 4 * Dc, Dc);

  // 8) MLP2 + bias + residual(x1) -> fp32 output
  gemm_bf16_kernel<true, false, true, false>
      <<<dim3(Dc / 128, Mc / 128), 256, 0, stream>>>(
          h1, w2T, b2, x1, out, Mc, Dc, 4 * Dc);
}
